// ConditionalQuadratic_8598524526718
// MI455X (gfx1250) — matrix operand order repaired and checked by reading, not compiled, not run on hardware
//
#include <hip/hip_runtime.h>
#include <hip/hip_bf16.h>

typedef __attribute__((ext_vector_type(16))) __bf16 v16bf;
typedef __attribute__((ext_vector_type(8)))  float  v8f;
typedef __attribute__((ext_vector_type(4)))  float  f4;

#define BSZ   32768
#define COND  256
#define DOUT  512
#define KK    64
#define NSEG  63          // K-1 segments
#define ROWS  128         // rows per workgroup
#define NTHREADS 256
#define BF_STRIDE 264     // bf16 elems per LDS row (256 + 8 pad) -> 528 bytes (4-bank row shift)
#define ROW_BYTES 528
#define H_STRIDE  132     // f32 elems per h row (132*4 = 528 B, aliases A tile)
#define COEF_STRIDE 256   // 64 segs * 4 floats (a,b,c,pad) per row -> float4 gathers
#define MIN_F 0.01f
#define W_BYTES (128 * ROW_BYTES)   // 67584: bf16 W image (incl. pad cols + zero row 127)

// LDS byte offsets
#define OFF_W    0                       // [128][264] bf16  = 67584 B
#define OFF_A    67584                   // [128][264] bf16  = 67584 B (reused as h f32 [128][132])
#define OFF_COEF 135168                  // [128][256] f32   = 131072 B
#define OFF_PART 266240                  // [256] f32
#define OFF_INV  267264                  // [128] f32
#define SMEM_BYTES 267776

__device__ __forceinline__ float gelu_exact(float x) {
    return 0.5f * x * (1.0f + erff(x * 0.70710678118654752f));
}
__device__ __forceinline__ float softplus_f(float x) {
    return fmaxf(x, 0.0f) + log1pf(expf(-fabsf(x)));
}
__device__ __forceinline__ float quad_int(float x, float a, float b, float c) {
    return (a * (1.0f/3.0f)) * x * x * x + (b * 0.5f) * x * x + c * x;
}

union FragU { uint4 u[2]; v16bf v; };

// ---- prep: W_h (127x256 f32) -> bf16 image matching the LDS layout (stride 264, row 127 + pads = 0)
__global__ __launch_bounds__(NTHREADS) void cq_prep_w(
    const float* __restrict__ Wh, __bf16* __restrict__ Wpre)
{
    int idx = blockIdx.x * NTHREADS + threadIdx.x;          // 132 blocks -> 33792 elems
    int n = idx / BF_STRIDE;
    int k = idx - n * BF_STRIDE;
    float v = (n < 127 && k < COND) ? Wh[n * COND + k] : 0.0f;
    Wpre[idx] = (__bf16)v;
}

__global__ __launch_bounds__(NTHREADS) void cq_fused_kernel(
    const float* __restrict__ Z,     // [BSZ][COND]
    const float* __restrict__ Y,     // [BSZ][DOUT]
    const float* __restrict__ Wh,    // [127][256]
    const float* __restrict__ Bh,    // [127]
    const __bf16* __restrict__ Wpre, // bf16 LDS image in workspace (or null)
    int useWs,
    float* __restrict__ Out)         // [BSZ][DOUT]
{
    extern __shared__ char smem[];
    __bf16* smW   = (__bf16*)(smem + OFF_W);     // smW[n][k]
    __bf16* smA   = (__bf16*)(smem + OFF_A);     // smA[m][k]
    float*  smH   = (float*)(smem + OFF_A);      // aliases smA after GEMM: smH[m][n]
    float*  sCoef = (float*)(smem + OFF_COEF);   // sCoef[row][seg*4 + {a,b,c,pad}]
    float*  sPart = (float*)(smem + OFF_PART);
    float*  sInv  = (float*)(smem + OFF_INV);

    const int tid  = threadIdx.x;
    const int lane = tid & 31;
    const int wave = tid >> 5;
    const int rowbase = blockIdx.x * ROWS;

    // ---- Phase 0: stage W image into LDS ----
    if (useWs) {
        // CDNA5 async DMA: global -> LDS, bypasses VGPRs, overlaps with the gelu phase below.
        const unsigned ldsW = (unsigned)(uintptr_t)(smem + OFF_W);
        #pragma unroll 4
        for (int t = 0; t < W_BYTES / (8 * NTHREADS); ++t) {     // 33 iterations of b64
            int i = t * NTHREADS + tid;
            unsigned lo = ldsW + i * 8;
            unsigned long long ga = (unsigned long long)(const char*)Wpre + (unsigned long long)(i * 8);
            asm volatile("global_load_async_to_lds_b64 %0, %1, off"
                         :: "v"(lo), "v"(ga) : "memory");
        }
    } else {
        for (int i = tid; i < 127 * COND; i += NTHREADS) {
            int n = i >> 8, k = i & 255;
            smW[n * BF_STRIDE + k] = (__bf16)Wh[i];
        }
        smW[127 * BF_STRIDE + tid] = (__bf16)0.0f;
    }

    // ---- Phase 1: gelu(z) tile -> bf16 A tile (overlaps the async W copy) ----
    {
        const f4* z4 = (const f4*)(Z + (size_t)rowbase * COND);
        #pragma unroll 4
        for (int t = 0; t < (ROWS * COND) / (4 * NTHREADS); ++t) {  // 32 iters
            int i4 = t * NTHREADS + tid;
            f4 v = __builtin_nontemporal_load(&z4[i4]);             // z read exactly once
            int e = i4 * 4;
            int r = e >> 8;
            int k = e & 255;
            __bf16* dst = smA + r * BF_STRIDE + k;
            dst[0] = (__bf16)gelu_exact(v[0]);
            dst[1] = (__bf16)gelu_exact(v[1]);
            dst[2] = (__bf16)gelu_exact(v[2]);
            dst[3] = (__bf16)gelu_exact(v[3]);
        }
    }
    if (useWs) {
        asm volatile("s_wait_asynccnt 0x0" ::: "memory");
    }
    __syncthreads();

    // ---- Phase 2: WMMA GEMM. Each wave: 16 rows x 128 cols, K = 256 (8 steps of 32) ----
    // A frag (16-bit 16x32, ISA 7.12.2): lane m = lane&15, klo=(lane&16)?8:0;
    //   elems 0..7 -> K=klo..klo+7 ; elems 8..15 -> K=16+klo..23+klo
    // B frag (32x16): lane n = lane&15, klo=(lane&16)?8:0; elems 0..7 -> K=klo..klo+7 ; elems 8..15 -> K=16+klo..23+klo
    {
        const int am = wave * 16 + (lane & 15);
        const int akloB = ((lane & 16) ? 8 : 0) * 2;
        const char* aRow = (const char*)smA + am * ROW_BYTES;

        v16bf aF[8];
        #pragma unroll
        for (int kb = 0; kb < 8; ++kb) {
            FragU f;
            f.u[0] = *(const uint4*)(aRow + kb * 64 + akloB);
            f.u[1] = *(const uint4*)(aRow + kb * 64 + 32 + akloB);
            aF[kb] = f.v;
        }
        __syncthreads();   // all A reads done before h stores overwrite this region

        const int nlocal = lane & 15;
        const int bkB = (lane & 16) ? 16 : 0;
        const int mrow = wave * 16 + ((lane & 16) ? 8 : 0);

        #pragma unroll
        for (int ct = 0; ct < 8; ++ct) {
            const int n = ct * 16 + nlocal;
            const char* wRow = (const char*)smW + n * ROW_BYTES;
            const float bias = (n < 127) ? Bh[n] : 0.0f;

            // Load ALL B fragments first so the 8-deep WMMA chain isn't per-step load-stalled.
            v16bf bF[8];
            #pragma unroll
            for (int kb = 0; kb < 8; ++kb) {
                FragU f;
                f.u[0] = *(const uint4*)(wRow + kb * 64 + bkB);
                f.u[1] = *(const uint4*)(wRow + kb * 64 + bkB + 32);
                bF[kb] = f.v;
            }
            v8f acc = {};
            #pragma unroll
            for (int kb = 0; kb < 8; ++kb) {
                acc = __builtin_amdgcn_wmma_f32_16x16x32_bf16(
                        false, aF[kb], false, bF[kb], (short)0, acc, false, false);
            }
            #pragma unroll
            for (int r = 0; r < 8; ++r) {
                smH[(mrow + r) * H_STRIDE + n] = acc[r] + bias;
            }
        }
    }
    __syncthreads();

    // ---- Phase 2b: softplus the 64 knot columns in place (each knot done once, not twice) ----
    for (int i = tid; i < ROWS * KK; i += NTHREADS) {
        int r = i >> 6, k = i & 63;
        float* p = smH + r * H_STRIDE + k;
        *p = softplus_f(*p);
    }
    __syncthreads();

    // ---- Phase 3: per-row Lagrange coefficients + integral (2 threads per row) ----
    {
        const int row  = tid >> 1;
        const int half = tid & 1;
        const int s0 = half ? 32 : 0;
        const int s1 = half ? NSEG : 32;
        const float* hrow = smH + row * H_STRIDE;
        float* crow = sCoef + row * COEF_STRIDE;
        const float w = 1.0f / (float)NSEG;
        float partial = 0.0f;

        for (int seg = s0; seg < s1; ++seg) {
            float x1 = (float)seg * w;
            float x3 = x1 + w;
            float x2 = x1 + 0.5f * w;
            float y1 = hrow[seg];          // softplus'd knots
            float y3 = hrow[seg + 1];
            float y2 = hrow[KK + seg];

            float denom = (x1 - x2) * (x1 - x3) * (x2 - x3);
            float inv_d = 1.0f / denom;
            float a = (x3 * (y2 - y1) + x2 * (y1 - y3) + x1 * (y3 - y2)) * inv_d;
            float b = (x1 * x1 * (y2 - y3) + x3 * x3 * (y1 - y2) + x2 * x2 * (y3 - y1)) * inv_d;
            float c = (x2 * x2 * (x3 * y1 - x1 * y3) + x2 * (x1 * x1 * y3 - x3 * x3 * y1)
                       + x1 * x3 * (x3 - x1) * y2) * inv_d;
            f4 co; co[0] = a; co[1] = b; co[2] = c; co[3] = 0.0f;
            *(f4*)(crow + seg * 4) = co;                        // one ds_store_b128

            float above = quad_int(x3, a, b, c) - quad_int(x1, a, b, c);
            float disc = b * b - 4.0f * a * (c - MIN_F);
            float sgn = (a > 0.0f) ? 1.0f : ((a < 0.0f) ? -1.0f : 0.0f);
            float root_denom = fmaxf(fabsf(2.0f * a), 1e-10f) * (sgn + 1e-10f);
            float inv_rd = 1.0f / root_denom;                   // shared by r1,r2
            float sq = sqrtf(fmaxf(disc, 1e-10f));
            float r1 = (-b - sq) * inv_rd;
            float r2 = (-b + sq) * inv_rd;
            float sr = fminf(r1, r2);
            float br = fmaxf(r1, r2);
            sr = fminf(fmaxf(sr, x1), x3);
            br = fmaxf(fminf(br, x3), x1);
            float between = quad_int(br, a, b, c) - quad_int(sr, a, b, c);
            float parab_max = c - MIN_F - b * b / (4.0f * a);
            bool ignore = (disc <= 0.0f) | (parab_max > 0.0f) | isnan(r1) | isnan(r2);
            between = ignore ? 0.0f : between;
            float min_int = ignore ? 0.0f : (br - sr) * MIN_F;
            float fint = above - between + min_int;
            partial += fmaxf(fint, 0.001f);
        }
        sPart[tid] = partial;
    }
    __syncthreads();
    if (tid < ROWS) {
        sInv[tid] = 1.0f / (sPart[2 * tid] + sPart[2 * tid + 1]);
    }
    __syncthreads();

    // ---- Phase 4: evaluate 512 query points per row; stream y -> out (nontemporal) ----
    {
        const f4* y4 = (const f4*)(Y + (size_t)rowbase * DOUT);
        f4* o4 = (f4*)(Out + (size_t)rowbase * DOUT);
        #pragma unroll 4
        for (int t = 0; t < (ROWS * DOUT) / (4 * NTHREADS); ++t) {   // 64 iters
            int i4 = t * NTHREADS + tid;
            f4 yv = __builtin_nontemporal_load(&y4[i4]);
            int e = i4 * 4;
            int row = e >> 9;              // DOUT = 512
            float inv = sInv[row];
            const float* crow = sCoef + row * COEF_STRIDE;
            f4 ov;
            #pragma unroll
            for (int q = 0; q < 4; ++q) {
                float yc = fminf(fmaxf(yv[q], 0.0f), 1.0f - 1e-5f);
                int idx = (int)(yc * (float)NSEG);
                idx = min(idx, NSEG - 1);                       // searchsorted on uniform grid
                f4 co = *(const f4*)(crow + idx * 4);           // one ds_load_b128
                float v = (co[0] * yc * yc + co[1] * yc + co[2]) * inv;
                ov[q] = fmaxf(v, MIN_F);
            }
            __builtin_nontemporal_store(ov, &o4[i4]);
        }
    }
}

extern "C" void kernel_launch(void* const* d_in, const int* in_sizes, int n_in,
                              void* d_out, int out_size, void* d_ws, size_t ws_size,
                              hipStream_t stream) {
    const float* z  = (const float*)d_in[0];
    const float* y  = (const float*)d_in[1];
    const float* Wh = (const float*)d_in[2];
    const float* Bh = (const float*)d_in[3];
    float* out = (float*)d_out;
    (void)in_sizes; (void)n_in; (void)out_size;

    const int useWs = (d_ws != nullptr && ws_size >= (size_t)W_BYTES) ? 1 : 0;
    if (useWs) {
        cq_prep_w<<<dim3((128 * BF_STRIDE) / NTHREADS), dim3(NTHREADS), 0, stream>>>(
            Wh, (__bf16*)d_ws);
    }
    cq_fused_kernel<<<dim3(BSZ / ROWS), dim3(NTHREADS), SMEM_BYTES, stream>>>(
        z, y, Wh, Bh, (const __bf16*)d_ws, useWs, out);
}
